// BitDeltaLinear_40896678593009
// MI455X (gfx1250) — matrix operand order repaired and checked by reading, not compiled, not run on hardware
//
#include <hip/hip_runtime.h>

// ---------------------------------------------------------------------------
// BitDelta linear:  out[M,N] = A[M,K] * (base + sign*scale)[N,K]^T
//   M = 16384, N = 4096, K = 4096  -> 549 GFLOP, compute-bound.
//   1) prep kernel: W_bf16 into d_ws (32 MB), once per call.
//   2) GEMM: v_wmma_f32_16x16x32_bf16, double-buffered LDS ping-pong:
//      - W slab kt+1 via GLOBAL_LOAD_ASYNC_TO_LDS_B128 (ASYNCcnt) issued
//        before compute -> DMA hides under 16 WMMAs.
//      - A slab kt+1 global->regs before compute; cvt+ds_store after compute.
//      - one s_barrier per K-step.
//   3) Fallback fused kernel if ws_size < 32 MB.
// ---------------------------------------------------------------------------

typedef __attribute__((ext_vector_type(16))) __bf16  v16bf;
typedef __attribute__((ext_vector_type(8)))  __bf16  bf16x8;
typedef __attribute__((ext_vector_type(8)))  float   v8f;

constexpr int GM = 8 * 2048;  // 16384
constexpr int GK = 4096;
constexpr int GN = 4096;

// ===========================================================================
// Kernel 1: materialize bf16 weight  W[o][i] = bf16(base + (s ? +sc : -sc))
// ===========================================================================
__global__ __launch_bounds__(256)
void build_w_kernel(const float* __restrict__ Wbase,
                    const int*   __restrict__ Wsign,
                    const float* __restrict__ Wscale,
                    unsigned short* __restrict__ Wb16)
{
    const long long p = (long long)blockIdx.x * 256 + threadIdx.x; // chunk of 8
    const int row = (int)(p >> 9);          // GK/8 = 512 chunks per row
    const int c8  = ((int)p & 511) * 8;
    const size_t off = (size_t)row * GK + c8;

    const float4* wb = reinterpret_cast<const float4*>(Wbase + off);
    const int4*   sg = reinterpret_cast<const int4*>(Wsign + off);
    const float   sc = Wscale[row];
    float4 b0 = wb[0], b1 = wb[1];
    int4   s0 = sg[0], s1 = sg[1];
    bf16x8 h;
    h[0] = (__bf16)(b0.x + (s0.x ? sc : -sc));
    h[1] = (__bf16)(b0.y + (s0.y ? sc : -sc));
    h[2] = (__bf16)(b0.z + (s0.z ? sc : -sc));
    h[3] = (__bf16)(b0.w + (s0.w ? sc : -sc));
    h[4] = (__bf16)(b1.x + (s1.x ? sc : -sc));
    h[5] = (__bf16)(b1.y + (s1.y ? sc : -sc));
    h[6] = (__bf16)(b1.z + (s1.z ? sc : -sc));
    h[7] = (__bf16)(b1.w + (s1.w ? sc : -sc));
    *reinterpret_cast<bf16x8*>(Wb16 + off) = h;
}

// ===========================================================================
// Kernel 2: main GEMM. Block tile 128(M) x 256(N), BK=32, double-buffered.
// 8 waves: 2 along M x 4 along N, each wave owns a 64x64 tile (4x4 WMMA).
// ===========================================================================
constexpr int BM   = 128;
constexpr int BN   = 256;
constexpr int BKT  = 32;        // one WMMA K-step per barrier
constexpr int LDSS = BKT + 8;   // 40 shorts = 80 B stride (16B aligned)

__global__ __launch_bounds__(256)
void bitdelta_gemm_kernel(const float*          __restrict__ A,    // [GM][GK] fp32
                          const unsigned short* __restrict__ Wb16, // [GN][GK] bf16
                          float*                __restrict__ Out)  // [GM][GN] fp32
{
    __shared__ __align__(16) unsigned short lds_a[2][BM][LDSS];  // 20 KB
    __shared__ __align__(16) unsigned short lds_b[2][BN][LDSS];  // 40 KB

    const int tid    = threadIdx.x;
    const int lane   = tid & 31;
    const int wave   = tid >> 5;     // 0..7
    const int wave_m = wave & 1;     // 2 waves along M, 64 rows each
    const int wave_n = wave >> 1;    // 4 waves along N, 64 cols each
    const int half   = lane >> 4;
    const int lid    = lane & 15;

    const int m0 = blockIdx.y * BM;
    const int n0 = blockIdx.x * BN;

    // --- staging helpers -------------------------------------------------
    auto issueW = [&](int kb, int buf) {             // async DMA, no VGPR data
        #pragma unroll
        for (int it = 0; it < 4; ++it) {
            const int p   = tid + it * 256;          // 0..1023
            const int row = p >> 2;                  // 0..255
            const int c8  = (p & 3) * 8;             // 0..24
            const unsigned long long g =
                (unsigned long long)(Wb16 + (size_t)(n0 + row) * GK + kb + c8);
            const unsigned lds_off =
                (unsigned)(unsigned long long)&lds_b[buf][row][c8];
            asm volatile("global_load_async_to_lds_b128 %0, %1, off"
                         :: "v"(lds_off), "v"(g) : "memory");
        }
    };
    auto loadA = [&](int kb, float4 (&ar)[2][2]) {   // global -> regs
        #pragma unroll
        for (int it = 0; it < 2; ++it) {
            const int p   = tid + it * 256;          // 0..511
            const int row = p >> 2;                  // 0..127
            const int c8  = (p & 3) * 8;
            const float4* src = reinterpret_cast<const float4*>(
                A + (size_t)(m0 + row) * GK + kb + c8);
            ar[it][0] = src[0];
            ar[it][1] = src[1];
        }
    };
    auto storeA = [&](float4 (&ar)[2][2], int buf) { // cvt bf16 + ds_store
        #pragma unroll
        for (int it = 0; it < 2; ++it) {
            const int p   = tid + it * 256;
            const int row = p >> 2;
            const int c8  = (p & 3) * 8;
            bf16x8 h;
            h[0] = (__bf16)ar[it][0].x; h[1] = (__bf16)ar[it][0].y;
            h[2] = (__bf16)ar[it][0].z; h[3] = (__bf16)ar[it][0].w;
            h[4] = (__bf16)ar[it][1].x; h[5] = (__bf16)ar[it][1].y;
            h[6] = (__bf16)ar[it][1].z; h[7] = (__bf16)ar[it][1].w;
            *reinterpret_cast<bf16x8*>(&lds_a[buf][row][c8]) = h;
        }
    };
    // ---------------------------------------------------------------------

    v8f acc[4][4];
    #pragma unroll
    for (int i = 0; i < 4; ++i)
        #pragma unroll
        for (int j = 0; j < 4; ++j)
            acc[i][j] = (v8f){0.f, 0.f, 0.f, 0.f, 0.f, 0.f, 0.f, 0.f};

    const int nkt = GK / BKT;        // 128 K-steps

    // prologue: stage slab 0 into buffer 0
    float4 areg[2][2];
    issueW(0, 0);
    loadA(0, areg);
    storeA(areg, 0);
    asm volatile("s_wait_asynccnt 0" ::: "memory");
    __syncthreads();

    for (int kt = 0; kt < nkt; ++kt) {
        const int  buf = kt & 1;
        const bool nxt = (kt + 1) < nkt;

        // stage slab kt+1 into the other buffer; DMA + loads overlap compute
        if (nxt) {
            issueW((kt + 1) * BKT, buf ^ 1);
            loadA((kt + 1) * BKT, areg);
            __builtin_prefetch(A + (size_t)(m0 + (tid & 127)) * GK
                                 + (kt + 2) * BKT, 0, 1);
        }

        // ---- 16 WMMAs from current buffer ----
        union FragU { uint4 u[2]; v16bf v; };

        v16bf afrag[4];
        #pragma unroll
        for (int i = 0; i < 4; ++i) {
            // A 16x32 bf16 layout: lanes 0-15 -> K {0..7,16..23};
            //                      lanes 16-31 -> K {8..15,24..31}
            const int r  = wave_m * 64 + i * 16 + lid;
            const int kA = half * 8;
            FragU fa;
            fa.u[0] = *reinterpret_cast<const uint4*>(&lds_a[buf][r][kA]);
            fa.u[1] = *reinterpret_cast<const uint4*>(&lds_a[buf][r][kA + 16]);
            afrag[i] = fa.v;
        }

        v16bf bfrag[4];
        #pragma unroll
        for (int j = 0; j < 4; ++j) {
            // B 32x16 bf16 layout: lanes 0-15 -> K {0..7,16..23}; lanes 16-31 -> K {8..15,24..31}
            const int c  = wave_n * 64 + j * 16 + lid;
            const int kB = half * 8;
            FragU fb;
            fb.u[0] = *reinterpret_cast<const uint4*>(&lds_b[buf][c][kB]);
            fb.u[1] = *reinterpret_cast<const uint4*>(&lds_b[buf][c][kB + 16]);
            bfrag[j] = fb.v;
        }

        #pragma unroll
        for (int i = 0; i < 4; ++i)
            #pragma unroll
            for (int j = 0; j < 4; ++j)
                acc[i][j] = __builtin_amdgcn_wmma_f32_16x16x32_bf16(
                    false, afrag[i], false, bfrag[j],
                    (short)0, acc[i][j], false, false);

        // A regs -> other LDS buffer (s_wait_loadcnt sinks here, after WMMAs)
        if (nxt) storeA(areg, buf ^ 1);

        asm volatile("s_wait_asynccnt 0" ::: "memory"); // W DMA for kt+1 landed
        __syncthreads();                                // buffer swap point
    }

    // ---- epilogue: f32 C/D layout: VGPR r -> M=r (lanes 0-15) / M=r+8 ----
    #pragma unroll
    for (int i = 0; i < 4; ++i) {
        const int mb = m0 + wave_m * 64 + i * 16 + half * 8;
        #pragma unroll
        for (int j = 0; j < 4; ++j) {
            const int n = n0 + wave_n * 64 + j * 16 + lid;
            #pragma unroll
            for (int r = 0; r < 8; ++r)
                Out[(size_t)(mb + r) * GN + n] = acc[i][j][r];
        }
    }
}

// ===========================================================================
// Kernel 3: fallback fully-fused kernel if ws too small.
// ===========================================================================
constexpr int FBM = 128, FBN = 128, FBKT = 64, FLDSS = FBKT + 8;

__global__ __launch_bounds__(256)
void bitdelta_fused_kernel(const float* __restrict__ A,
                           const float* __restrict__ Wbase,
                           const int*   __restrict__ Wsign,
                           const float* __restrict__ Wscale,
                           float*       __restrict__ Out)
{
    __shared__ __align__(16) unsigned short lds_a[FBM][FLDSS];
    __shared__ __align__(16) unsigned short lds_b[FBN][FLDSS];

    const int tid    = threadIdx.x;
    const int lane   = tid & 31;
    const int wave   = tid >> 5;
    const int wave_m = wave & 3;
    const int wave_n = wave >> 2;
    const int half   = lane >> 4;
    const int lid    = lane & 15;

    const int m0 = blockIdx.y * FBM;
    const int n0 = blockIdx.x * FBN;

    v8f acc[2][4];
    #pragma unroll
    for (int i = 0; i < 2; ++i)
        #pragma unroll
        for (int j = 0; j < 4; ++j)
            acc[i][j] = (v8f){0.f, 0.f, 0.f, 0.f, 0.f, 0.f, 0.f, 0.f};

    const int nkt = GK / FBKT;

    for (int kt = 0; kt < nkt; ++kt) {
        const int kb = kt * FBKT;
        __syncthreads();

        #pragma unroll
        for (int it = 0; it < 4; ++it) {
            const int p   = tid + it * 256;
            const int row = p >> 3;
            const int c8  = (p & 7) * 8;
            {
                const float4* src = reinterpret_cast<const float4*>(
                    A + (size_t)(m0 + row) * GK + kb + c8);
                float4 f0 = src[0], f1 = src[1];
                bf16x8 h;
                h[0] = (__bf16)f0.x; h[1] = (__bf16)f0.y;
                h[2] = (__bf16)f0.z; h[3] = (__bf16)f0.w;
                h[4] = (__bf16)f1.x; h[5] = (__bf16)f1.y;
                h[6] = (__bf16)f1.z; h[7] = (__bf16)f1.w;
                *reinterpret_cast<bf16x8*>(&lds_a[row][c8]) = h;
            }
            {
                const size_t off = (size_t)(n0 + row) * GK + kb + c8;
                const float4* wb = reinterpret_cast<const float4*>(Wbase + off);
                const int4*   sg = reinterpret_cast<const int4*>(Wsign + off);
                const float   sc = Wscale[n0 + row];
                float4 b0 = wb[0], b1 = wb[1];
                int4   s0 = sg[0], s1 = sg[1];
                bf16x8 h;
                h[0] = (__bf16)(b0.x + (s0.x ? sc : -sc));
                h[1] = (__bf16)(b0.y + (s0.y ? sc : -sc));
                h[2] = (__bf16)(b0.z + (s0.z ? sc : -sc));
                h[3] = (__bf16)(b0.w + (s0.w ? sc : -sc));
                h[4] = (__bf16)(b1.x + (s1.x ? sc : -sc));
                h[5] = (__bf16)(b1.y + (s1.y ? sc : -sc));
                h[6] = (__bf16)(b1.z + (s1.z ? sc : -sc));
                h[7] = (__bf16)(b1.w + (s1.w ? sc : -sc));
                *reinterpret_cast<bf16x8*>(&lds_b[row][c8]) = h;
            }
        }
        __syncthreads();

        union FragU { uint4 u[2]; v16bf v; };
        #pragma unroll
        for (int ks = 0; ks < 2; ++ks) {
            const int kf = ks * 32;
            v16bf afrag[2];
            #pragma unroll
            for (int i = 0; i < 2; ++i) {
                const int r  = wave_m * 32 + i * 16 + lid;
                const int kA = kf + half * 8;
                FragU fa;
                fa.u[0] = *reinterpret_cast<const uint4*>(&lds_a[r][kA]);
                fa.u[1] = *reinterpret_cast<const uint4*>(&lds_a[r][kA + 16]);
                afrag[i] = fa.v;
            }
            v16bf bfrag[4];
            #pragma unroll
            for (int j = 0; j < 4; ++j) {
                const int c  = wave_n * 64 + j * 16 + lid;
                const int kB = kf + half * 8;
                FragU fb;
                fb.u[0] = *reinterpret_cast<const uint4*>(&lds_b[c][kB]);
                fb.u[1] = *reinterpret_cast<const uint4*>(&lds_b[c][kB + 16]);
                bfrag[j] = fb.v;
            }
            #pragma unroll
            for (int i = 0; i < 2; ++i)
                #pragma unroll
                for (int j = 0; j < 4; ++j)
                    acc[i][j] = __builtin_amdgcn_wmma_f32_16x16x32_bf16(
                        false, afrag[i], false, bfrag[j],
                        (short)0, acc[i][j], false, false);
        }
    }

    #pragma unroll
    for (int i = 0; i < 2; ++i) {
        const int mb = m0 + wave_m * 32 + i * 16 + half * 8;
        #pragma unroll
        for (int j = 0; j < 4; ++j) {
            const int n = n0 + wave_n * 64 + j * 16 + lid;
            #pragma unroll
            for (int r = 0; r < 8; ++r)
                Out[(size_t)(mb + r) * GN + n] = acc[i][j][r];
        }
    }
}

// ===========================================================================
extern "C" void kernel_launch(void* const* d_in, const int* in_sizes, int n_in,
                              void* d_out, int out_size, void* d_ws, size_t ws_size,
                              hipStream_t stream) {
    const float* A      = (const float*)d_in[0];
    const float* Wbase  = (const float*)d_in[1];
    const int*   Wsign  = (const int*)  d_in[2];
    const float* Wscale = (const float*)d_in[3];
    float*       Out    = (float*)d_out;

    const size_t w_bytes = (size_t)GN * GK * sizeof(unsigned short); // 32 MB

    if (ws_size >= w_bytes) {
        unsigned short* Wb16 = (unsigned short*)d_ws;
        // 1) materialize bf16 weight: GN*GK/8 chunks / 256 threads = 8192 blocks
        build_w_kernel<<<8192, 256, 0, stream>>>(Wbase, Wsign, Wscale, Wb16);
        // 2) GEMM
        dim3 grid(GN / BN, GM / BM);       // 16 x 128
        bitdelta_gemm_kernel<<<grid, 256, 0, stream>>>(A, Wb16, Out);
    } else {
        dim3 grid(GN / FBN, GM / FBM);     // 32 x 128
        bitdelta_fused_kernel<<<grid, 256, 0, stream>>>(A, Wbase, Wsign, Wscale, Out);
    }
}
